// TopologyAwareAttention_22917945491604
// MI455X (gfx1250) — compile-verified, matrix operand order checked by reading, not run on hardware
//
#include <hip/hip_runtime.h>

// ---------------------------------------------------------------------------
// TopologyAwareAttention on gfx1250 (MI455X), wave32.
// v_wmma_f32_16x16x32_f16 for all matmuls, global_load_async_to_lds_b128 for
// tile movement (double-buffered), DPP16 row reductions for online softmax.
// ---------------------------------------------------------------------------

typedef _Float16 f16;
typedef __attribute__((ext_vector_type(8)))  _Float16 f16x8;
typedef __attribute__((ext_vector_type(16))) _Float16 v16h;
typedef __attribute__((ext_vector_type(8)))  float    v8f;
typedef int v4i __attribute__((vector_size(16)));   // builtin's pointee type

#define BB 2
#define NN 2048
#define CC 512
#define HH 8
#define DH 64

#if defined(__AMDGCN__) && __has_builtin(__builtin_amdgcn_global_load_async_to_lds_b128)
#define HAS_ASYNC 1
#else
#define HAS_ASYNC 0
#endif

#if HAS_ASYNC
// async copy of 16 bytes global -> LDS (tracked by ASYNCcnt, no VGPR staging)
__device__ inline void async_cp16(const f16* g, f16* l) {
  auto gp = (__attribute__((address_space(1))) v4i*)(v4i*)(void*)g;
  auto lp = (__attribute__((address_space(3))) v4i*)(v4i*)l;
  __builtin_amdgcn_global_load_async_to_lds_b128(gp, lp, 0, 0);
}
__device__ inline void wait_async() {
#if __has_builtin(__builtin_amdgcn_s_wait_asynccnt)
  __builtin_amdgcn_s_wait_asynccnt(0);
#else
  asm volatile("s_wait_asynccnt 0x0" ::: "memory");
#endif
}
#endif

// Build a 16-half A/B fragment from a row-major LDS row pointer.
// Lane L: koff = (L>>4)*8 ; halves[0..7] = p[koff..], halves[8..15] = p[16+koff..]
__device__ inline v16h ldfrag(const f16* p, int koff) {
  f16x8 lo = *(const f16x8*)(p + koff);
  f16x8 hi = *(const f16x8*)(p + 16 + koff);
  return __builtin_shufflevector(lo, hi, 0,1,2,3,4,5,6,7,8,9,10,11,12,13,14,15);
}

__device__ inline v8f wmma_f16(v16h a, v16h b, v8f c) {
  return __builtin_amdgcn_wmma_f32_16x16x32_f16(false, a, false, b,
                                                (short)0, c, false, false);
}

// ---- 16-lane row reductions (wave32): DPP16 butterfly, no LDS traffic ----
#if defined(__AMDGCN__) && __has_builtin(__builtin_amdgcn_update_dpp)
#define ROW16_STEP(v, ctrl) \
  __int_as_float(__builtin_amdgcn_update_dpp(0, __float_as_int(v), ctrl, 0xf, 0xf, true))
__device__ inline float row16_max(float x) {
  x = fmaxf(x, ROW16_STEP(x, 0xB1));   // quad_perm [1,0,3,2]  (xor 1)
  x = fmaxf(x, ROW16_STEP(x, 0x4E));   // quad_perm [2,3,0,1]  (xor 2)
  x = fmaxf(x, ROW16_STEP(x, 0x141));  // row_half_mirror      (other quad in 8)
  x = fmaxf(x, ROW16_STEP(x, 0x140));  // row_mirror           (other 8 in 16)
  return x;
}
__device__ inline float row16_sum(float x) {
  x += ROW16_STEP(x, 0xB1);
  x += ROW16_STEP(x, 0x4E);
  x += ROW16_STEP(x, 0x141);
  x += ROW16_STEP(x, 0x140);
  return x;
}
#else
__device__ inline float row16_max(float x) {
  x = fmaxf(x, __shfl_xor(x, 1)); x = fmaxf(x, __shfl_xor(x, 2));
  x = fmaxf(x, __shfl_xor(x, 4)); x = fmaxf(x, __shfl_xor(x, 8));
  return x;
}
__device__ inline float row16_sum(float x) {
  x += __shfl_xor(x, 1); x += __shfl_xor(x, 2);
  x += __shfl_xor(x, 4); x += __shfl_xor(x, 8);
  return x;
}
#endif

// LCP bucket: matches reference (MAX_DEPTH=14, 16 buckets, trunc toward zero)
__device__ inline int lcp_bucket(int a, int b) {
  unsigned x = (unsigned)(a ^ b);
  if (x == 0u) return 15;
  int db  = 31 - __clz((int)x);
  int lcp = 14 - db / 3;
  int bu  = (int)((float)lcp * (15.0f / 14.0f));
  bu = bu < 0 ? 0 : bu;
  return bu > 15 ? 15 : bu;
}

// ---------------------------------------------------------------------------
// 128x128-tile WMMA GEMM:  Out[M,Nout] = A[M,K] * W[Nout,K]^T + bias
// Double-buffered LDS, async global->LDS tile copies, 1 barrier per k-step.
// MODE 0: f16 out (QKV proj) | MODE 1: conv-as-GEMM, scatter to qkv2 |
// MODE 2: f32 out (final projection)
// ---------------------------------------------------------------------------
template<int MODE>
__global__ __launch_bounds__(256)
void gemm_wmma(const f16* __restrict__ A, const f16* __restrict__ W,
               const float* __restrict__ bias,
               f16* __restrict__ outH, float* __restrict__ outF,
               int Nout, int Kt /* K-steps of 32 */)
{
  __shared__ __attribute__((aligned(16))) f16 sA[2][128][40];
  __shared__ __attribute__((aligned(16))) f16 sB[2][128][40];

  const int tid  = threadIdx.x;
  const int lane = tid & 31;
  const int wid  = tid >> 5;            // 8 waves: 4 along M x 2 along N
  const int wm   = wid & 3;
  const int wn   = wid >> 2;
  const int lm   = lane & 15;
  const int lh   = lane >> 4;
  const int koff = lh * 8;

  const int rowBase = blockIdx.x * 128;
  const int colBase = blockIdx.y * 128;
  const int K = Kt * 32;                // row stride for MODE 0/2

  int sIdx = 0, bIdx = 0;
  const f16* Abase = A;
  if (MODE == 1) {
    sIdx  = blockIdx.z / BB;
    bIdx  = blockIdx.z % BB;
    Abase = A + (size_t)bIdx * NN * (3 * CC) + (size_t)sIdx * CC;
  }

  v8f acc[2][4];
  for (int nt = 0; nt < 4; ++nt) {
    float bv = bias[colBase + wn * 64 + nt * 16 + lm];
    v8f c;
    for (int i = 0; i < 8; ++i) c[i] = bv;
    acc[0][nt] = c;
    acc[1][nt] = c;
  }

  const int arow = tid >> 2;            // 0..63 (+64 second chunk)
  const int acol = (tid & 3) * 8;

#if HAS_ASYNC
  auto issue_tile = [&](int ks, int buf) {
    if (MODE == 1) {
      int t = ks >> 4, ci0 = (ks & 15) * 32;
      for (int i = 0; i < 2; ++i) {
        int g = rowBase + arow + 64 * i + t - 2;         // shifted seq row
        if (g >= 0 && g < NN) {
          async_cp16(Abase + (size_t)g * (3 * CC) + ci0 + acol,
                     &sA[buf][arow + 64 * i][acol]);
        } else {
          f16x8 z; for (int q = 0; q < 8; ++q) z[q] = (_Float16)0;
          *(f16x8*)&sA[buf][arow + 64 * i][acol] = z;    // zero padding
        }
        async_cp16(W + ((size_t)t * CC + colBase + arow + 64 * i) * CC + ci0 + acol,
                   &sB[buf][arow + 64 * i][acol]);
      }
    } else {
      int k0 = ks * 32;
      for (int i = 0; i < 2; ++i) {
        async_cp16(A + (size_t)(rowBase + arow + 64 * i) * K + k0 + acol,
                   &sA[buf][arow + 64 * i][acol]);
        async_cp16(W + (size_t)(colBase + arow + 64 * i) * K + k0 + acol,
                   &sB[buf][arow + 64 * i][acol]);
      }
    }
  };
  issue_tile(0, 0);
  wait_async();
  __syncthreads();
#else
  f16x8 ra[2], rb[2];
  auto load_global = [&](int ks) {
    if (MODE == 1) {
      int t = ks >> 4, ci0 = (ks & 15) * 32;
      for (int i = 0; i < 2; ++i) {
        int g = rowBase + arow + 64 * i + t - 2;
        if (g >= 0 && g < NN) {
          ra[i] = *(const f16x8*)(Abase + (size_t)g * (3 * CC) + ci0 + acol);
        } else {
          f16x8 z; for (int q = 0; q < 8; ++q) z[q] = (_Float16)0; ra[i] = z;
        }
        rb[i] = *(const f16x8*)(W + ((size_t)t * CC + colBase + arow + 64 * i) * CC + ci0 + acol);
      }
    } else {
      int k0 = ks * 32;
      for (int i = 0; i < 2; ++i) {
        ra[i] = *(const f16x8*)(A + (size_t)(rowBase + arow + 64 * i) * K + k0 + acol);
        rb[i] = *(const f16x8*)(W + (size_t)(colBase + arow + 64 * i) * K + k0 + acol);
      }
    }
  };
  auto store_lds = [&](int buf) {
    *(f16x8*)&sA[buf][arow     ][acol] = ra[0];
    *(f16x8*)&sA[buf][arow + 64][acol] = ra[1];
    *(f16x8*)&sB[buf][arow     ][acol] = rb[0];
    *(f16x8*)&sB[buf][arow + 64][acol] = rb[1];
  };
  load_global(0);
  store_lds(0);
  __syncthreads();
#endif

  int buf = 0;
  for (int ks = 0; ks < Kt; ++ks) {
#if HAS_ASYNC
    if (ks + 1 < Kt) issue_tile(ks + 1, buf ^ 1);   // overlap copy with WMMA
#else
    if (ks + 1 < Kt) load_global(ks + 1);
#endif

    v16h af[2], bf[4];
    for (int mt = 0; mt < 2; ++mt)
      af[mt] = ldfrag(&sA[buf][wm * 32 + mt * 16 + lm][0], koff);
    for (int nt = 0; nt < 4; ++nt)
      bf[nt] = ldfrag(&sB[buf][wn * 64 + nt * 16 + lm][0], koff);

    for (int mt = 0; mt < 2; ++mt)
      for (int nt = 0; nt < 4; ++nt)
        acc[mt][nt] = wmma_f16(af[mt], bf[nt], acc[mt][nt]);

#if HAS_ASYNC
    if (ks + 1 < Kt) wait_async();
#else
    if (ks + 1 < Kt) store_lds(buf ^ 1);
#endif
    __syncthreads();
    buf ^= 1;
  }

  // epilogue: D layout lane L -> col = lm, rows = lh*8 + r
  for (int mt = 0; mt < 2; ++mt) {
    for (int nt = 0; nt < 4; ++nt) {
      int col = colBase + wn * 64 + nt * 16 + lm;
      for (int r = 0; r < 8; ++r) {
        int row = rowBase + wm * 32 + mt * 16 + lh * 8 + r;
        float v = acc[mt][nt][r];
        if (MODE == 0) {
          outH[(size_t)row * Nout + col] = (f16)v;
        } else if (MODE == 1) {
          int h = col >> 6, dh = col & 63;
          outH[((((size_t)sIdx * BB + bIdx) * HH + h) * NN + row) * DH + dh] = (f16)v;
        } else {
          outF[(size_t)row * Nout + col] = v;
        }
      }
    }
  }
}

// ---------------------------------------------------------------------------
// Flash attention with on-the-fly LCP bucket bias.
// 128 threads = 4 waves x 16 query rows; KV tiles of 32.
// Per iter per wave: 4 WMMAs for S (K=64), 4 WMMAs for PV (K=32).
// ---------------------------------------------------------------------------
__global__ __launch_bounds__(128)
void flash_attn(const f16* __restrict__ qkv2, const int* __restrict__ morton,
                const float* __restrict__ bias_table, f16* __restrict__ attnOut)
{
  __shared__ __attribute__((aligned(16))) f16 sK [32][72];
  __shared__ __attribute__((aligned(16))) f16 sVT[64][40];
  __shared__ __attribute__((aligned(16))) f16 sP [4][16][40];
  __shared__ int   sMort[32];
  __shared__ float sBias[16];

  const int tid  = threadIdx.x;
  const int lane = tid & 31;
  const int wid  = tid >> 5;
  const int lm   = lane & 15;
  const int lh   = lane >> 4;
  const int koff = lh * 8;

  const int bh = blockIdx.y;            // b*H + h
  const int b  = bh / HH;
  const int h  = bh % HH;
  const int qbase = blockIdx.x * 64 + wid * 16;

  const size_t plane = (size_t)BB * HH * NN * DH;
  const f16* Q = qkv2 +             (size_t)bh * NN * DH;
  const f16* K = qkv2 + plane     + (size_t)bh * NN * DH;
  const f16* V = qkv2 + 2 * plane + (size_t)bh * NN * DH;

  if (tid < 16) sBias[tid] = bias_table[h * 16 + tid];

  const f16* qrow = Q + (size_t)(qbase + lm) * DH;
  v16h q0 = ldfrag(qrow,      koff);    // dh 0..31
  v16h q1 = ldfrag(qrow + 32, koff);    // dh 32..63

  int mq[8];
  for (int r = 0; r < 8; ++r) mq[r] = morton[b * NN + qbase + lh * 8 + r];

  float mrun[8], lrun[8];
  for (int r = 0; r < 8; ++r) { mrun[r] = -3.0e38f; lrun[r] = 0.0f; }
  v8f o[4];
  for (int c = 0; c < 4; ++c) for (int i = 0; i < 8; ++i) o[c][i] = 0.0f;

  const float scale = 0.125f;           // DH^-0.5

  for (int j = 0; j < NN; j += 32) {
    __syncthreads();                    // prior PV reads done before overwrite
    // K tile 32x64: async global->LDS
    for (int i = 0; i < 2; ++i) {
      int c  = tid + 128 * i;
      int ky = c >> 3, d8 = (c & 7) * 8;
#if HAS_ASYNC
      async_cp16(K + (size_t)(j + ky) * DH + d8, &sK[ky][d8]);
#else
      *(f16x8*)&sK[ky][d8] = *(const f16x8*)(K + (size_t)(j + ky) * DH + d8);
#endif
    }
    // V tile, transposed into sVT[dh][key] (needs VGPR staging)
    for (int i = 0; i < 2; ++i) {
      int c  = tid + 128 * i;
      int ky = c >> 3, d8 = (c & 7) * 8;
      f16x8 v = *(const f16x8*)(V + (size_t)(j + ky) * DH + d8);
      for (int e = 0; e < 8; ++e) sVT[d8 + e][ky] = v[e];
    }
    if (tid < 32) sMort[tid] = morton[b * NN + j + tid];
#if HAS_ASYNC
    wait_async();
#endif
    __syncthreads();

    // S = Q * K^T for keys [j, j+32)
    v8f Sa, Sb;
    for (int i = 0; i < 8; ++i) { Sa[i] = 0.0f; Sb[i] = 0.0f; }
    {
      v16h kf;
      kf = ldfrag(&sK[lm     ][0],  koff); Sa = wmma_f16(q0, kf, Sa);
      kf = ldfrag(&sK[lm     ][32], koff); Sa = wmma_f16(q1, kf, Sa);
      kf = ldfrag(&sK[16 + lm][0],  koff); Sb = wmma_f16(q0, kf, Sb);
      kf = ldfrag(&sK[16 + lm][32], koff); Sb = wmma_f16(q1, kf, Sb);
    }

    // scale + LCP bucket bias
    const int mka = sMort[lm];
    const int mkb = sMort[16 + lm];
    for (int r = 0; r < 8; ++r) {
      Sa[r] = Sa[r] * scale + sBias[lcp_bucket(mq[r], mka)];
      Sb[r] = Sb[r] * scale + sBias[lcp_bucket(mq[r], mkb)];
    }

    // online softmax; row reductions via DPP16 (no LDS)
    float corr[8];
    for (int r = 0; r < 8; ++r) {
      float mx = row16_max(fmaxf(Sa[r], Sb[r]));
      float nm = fmaxf(mrun[r], mx);
      corr[r]  = __expf(mrun[r] - nm);
      float pa = __expf(Sa[r] - nm);
      float pb = __expf(Sb[r] - nm);
      Sa[r] = pa; Sb[r] = pb;
      lrun[r] = lrun[r] * corr[r] + row16_sum(pa + pb);
      mrun[r] = nm;
    }
    for (int c = 0; c < 4; ++c)
      for (int r = 0; r < 8; ++r) o[c][r] *= corr[r];

    // P: D layout -> A layout via per-wave LDS tile
    for (int r = 0; r < 8; ++r) {
      sP[wid][lh * 8 + r][lm]      = (f16)Sa[r];
      sP[wid][lh * 8 + r][16 + lm] = (f16)Sb[r];
    }
    __syncthreads();                    // uniform; makes sP/sVT visible

    v16h pf = ldfrag(&sP[wid][lm][0], koff);
    for (int c = 0; c < 4; ++c) {
      v16h vf = ldfrag(&sVT[c * 16 + lm][0], koff);
      o[c] = wmma_f16(pf, vf, o[c]);
    }
  }

  // epilogue: normalize and store f16 into [B][N][H*DH]
  for (int c = 0; c < 4; ++c) {
    for (int r = 0; r < 8; ++r) {
      int row = qbase + lh * 8 + r;
      float v = o[c][r] / lrun[r];
      attnOut[(size_t)(b * NN + row) * CC + h * DH + c * 16 + lm] = (f16)v;
    }
  }
}

// ---------------------------------------------------------------------------
// Prep kernels
// ---------------------------------------------------------------------------
__global__ void cvt_f16(const float* __restrict__ s, f16* __restrict__ d, int n) {
  int i = blockIdx.x * blockDim.x + threadIdx.x;
  if (i < n) d[i] = (f16)s[i];
}

// w_conv [C][C][5] f32 -> wconv_t [5][C][C] f16
__global__ void conv_w_prep(const float* __restrict__ w, f16* __restrict__ d) {
  int i = blockIdx.x * blockDim.x + threadIdx.x;
  if (i < 5 * CC * CC) {
    int t  = i / (CC * CC);
    int r  = i % (CC * CC);
    int co = r / CC;
    int ci = r % CC;
    d[i] = (f16)w[(co * CC + ci) * 5 + t];
  }
}

// ---------------------------------------------------------------------------
extern "C" void kernel_launch(void* const* d_in, const int* in_sizes, int n_in,
                              void* d_out, int out_size, void* d_ws, size_t ws_size,
                              hipStream_t stream) {
  (void)in_sizes; (void)n_in; (void)out_size; (void)ws_size;
  const float* x          = (const float*)d_in[0];
  const int*   morton     = (const int*)  d_in[1];
  const float* w_qkv      = (const float*)d_in[2];
  const float* b_qkv      = (const float*)d_in[3];
  const float* w_conv     = (const float*)d_in[4];
  const float* b_conv     = (const float*)d_in[5];
  const float* w_proj     = (const float*)d_in[6];
  const float* b_proj     = (const float*)d_in[7];
  const float* bias_table = (const float*)d_in[8];
  float* out = (float*)d_out;

  char* ws = (char*)d_ws;
  size_t o = 0;
  f16* xh     = (f16*)(ws + o); o += (size_t)BB * NN * CC * 2;
  f16* wqkvh  = (f16*)(ws + o); o += (size_t)3 * CC * CC * 2;
  f16* qkv1   = (f16*)(ws + o); o += (size_t)BB * NN * 3 * CC * 2;
  f16* wconvt = (f16*)(ws + o); o += (size_t)5 * CC * CC * 2;
  f16* qkv2   = (f16*)(ws + o); o += (size_t)3 * BB * HH * NN * DH * 2;
  f16* attnh  = (f16*)(ws + o); o += (size_t)BB * NN * CC * 2;
  f16* wprojh = (f16*)(ws + o); o += (size_t)CC * CC * 2;

  const int MN = BB * NN;

  cvt_f16<<<(MN * CC + 255) / 256, 256, 0, stream>>>(x, xh, MN * CC);
  cvt_f16<<<(3 * CC * CC + 255) / 256, 256, 0, stream>>>(w_qkv, wqkvh, 3 * CC * CC);
  cvt_f16<<<(CC * CC + 255) / 256, 256, 0, stream>>>(w_proj, wprojh, CC * CC);
  conv_w_prep<<<(5 * CC * CC + 255) / 256, 256, 0, stream>>>(w_conv, wconvt);

  // 1) qkv1[4096][1536] = x * w_qkv^T + b_qkv
  gemm_wmma<0><<<dim3(MN / 128, (3 * CC) / 128, 1), 256, 0, stream>>>(
      xh, wqkvh, b_qkv, qkv1, nullptr, 3 * CC, CC / 32);

  // 2) conv as 5 shifted GEMMs; scatter into qkv2[3][B][H][N][DH]
  gemm_wmma<1><<<dim3(NN / 128, CC / 128, 3 * BB), 256, 0, stream>>>(
      qkv1, wconvt, b_conv, qkv2, nullptr, CC, (5 * CC) / 32);

  // 3) flash attention with LCP bias
  flash_attn<<<dim3(NN / 64, BB * HH, 1), 128, 0, stream>>>(
      qkv2, morton, bias_table, attnh);

  // 4) out[4096][512] = attn * w_proj^T + b_proj (f32 out)
  gemm_wmma<2><<<dim3(MN / 128, CC / 128, 1), 256, 0, stream>>>(
      attnh, wprojh, b_proj, nullptr, out, CC, CC / 32);
}
